// NNUE_43490838839498
// MI455X (gfx1250) — 8-bit matrix operand order repaired and checked by reading, compiled, not run on hardware
//
#include <hip/hip_runtime.h>
#include <hip/hip_bf16.h>

typedef __attribute__((ext_vector_type(8))) int v8i;
typedef __attribute__((ext_vector_type(4))) int v4i;
typedef __attribute__((ext_vector_type(2))) int v2i;

#define BATCH 1024
#define NDIM  256
#define NFEAT 640
#define KTAB  (641*256)

static constexpr size_t OFF_PP   = 0;
static constexpr size_t OFF_ZROW = OFF_PP   + 1024*640;
static constexpr size_t OFF_JOBS = OFF_ZROW + 640;
static constexpr size_t OFF_CNT  = OFF_JOBS + (2048+16)*4;
static constexpr size_t OFF_CUR  = OFF_CNT  + 256;
static constexpr size_t OFF_OFFS = OFF_CUR  + 256;
static constexpr size_t OFF_TK   = OFF_OFFS + 256;
static constexpr size_t OFF_TO   = OFF_TK   + 768;
static constexpr size_t OFF_TC   = OFF_TO   + 768;
static constexpr size_t OFF_NT   = OFF_TC   + 768;
static constexpr size_t OFF_DUMP = OFF_NT   + 16;
static constexpr size_t OFF_RES  = OFF_DUMP + 1024;
static constexpr size_t OFF_X    = OFF_RES  + 2u*1024*256*4;
static constexpr size_t OFF_X1   = OFF_X    + 1024*512;
static constexpr size_t OFF_W1P  = OFF_X1   + 1024*32;
static constexpr size_t OFF_W2P  = OFF_W1P  + 32*512;
static constexpr size_t OFF_OWP  = OFF_W2P  + 32*64;
static constexpr size_t OFF_PSUM = OFF_OWP  + 32;
static constexpr size_t OFF_WLO  = OFF_PSUM + 16;
static constexpr size_t OFF_WHI  = OFF_WLO  + 64u*256*640;

__global__ void k_prep(const int* __restrict__ pp, const int* __restrict__ w1,
                       const int* __restrict__ w2, const int* __restrict__ ow,
                       unsigned char* __restrict__ ws) {
  int tid = blockIdx.x * blockDim.x + threadIdx.x;
  int stride = gridDim.x * blockDim.x;
  unsigned char* PP = ws + OFF_PP;
  for (int i = tid; i < BATCH * NFEAT; i += stride) PP[i] = pp[i] ? 1 : 0;
  char* w1p = (char*)(ws + OFF_W1P);
  for (int i = tid; i < 32 * 512; i += stride) w1p[i] = (char)w1[i];
  char* w2p = (char*)(ws + OFF_W2P);
  for (int i = tid; i < 32 * 64; i += stride) {
    int n = i >> 6, j = i & 63;
    w2p[i] = (j < 32) ? (char)w2[n * 32 + j] : (char)0;
  }
  char* owp = (char*)(ws + OFF_OWP);
  for (int i = tid; i < 32; i += stride) owp[i] = (char)ow[i];
  unsigned char* zr = ws + OFF_ZROW;
  for (int i = tid; i < NFEAT; i += stride) zr[i] = 0;
  unsigned* jobs = (unsigned*)(ws + OFF_JOBS);
  for (int i = tid; i < 16; i += stride) jobs[2048 + i] = 0;
  int* cnt = (int*)(ws + OFF_CNT);
  int* cur = (int*)(ws + OFF_CUR);
  for (int i = tid; i < 64; i += stride) { cnt[i] = 0; cur[i] = 0; }
  if (tid == 0) *(int*)(ws + OFF_PSUM) = 0;
}

__global__ void k_wsplit(const int* __restrict__ W, unsigned char* __restrict__ ws) {
  int idx = blockIdx.x * blockDim.x + threadIdx.x;
  if (idx >= 64 * 256 * 640) return;
  int f  = idx % 640;
  int kn = idx / 640;
  int n  = kn & 255;
  int k  = kn >> 8;
  int w  = W[k * KTAB + f * 256 + n];
  ((unsigned char*)(ws + OFF_WLO))[idx] = (unsigned char)(w & 0xFF);
  ((char*)(ws + OFF_WHI))[idx]          = (char)((w >> 8) & 0xFF);
}

__global__ void k_count(const int* __restrict__ kings, unsigned char* __restrict__ ws) {
  int j = blockIdx.x * blockDim.x + threadIdx.x;
  if (j >= 2 * BATCH) return;
  int k = kings[(j >> 1) * 2 + (j & 1)];
  atomicAdd((int*)(ws + OFF_CNT) + k, 1);
}

__global__ void k_scan(unsigned char* __restrict__ ws) {
  if (threadIdx.x != 0 || blockIdx.x != 0) return;
  const int* cnt = (const int*)(ws + OFF_CNT);
  int* offs = (int*)(ws + OFF_OFFS);
  int* tk = (int*)(ws + OFF_TK);
  int* to = (int*)(ws + OFF_TO);
  int* tc = (int*)(ws + OFF_TC);
  int off = 0, T = 0;
  for (int k = 0; k < 64; ++k) {
    offs[k] = off;
    int c = cnt[k];
    for (int t = 0; t < c; t += 16) {
      int rem = c - t;
      tk[T] = k; to[T] = off + t; tc[T] = rem < 16 ? rem : 16; ++T;
    }
    off += c;
  }
  *(int*)(ws + OFF_NT) = T;
}

__global__ void k_scatter(const int* __restrict__ kings, unsigned char* __restrict__ ws) {
  int j = blockIdx.x * blockDim.x + threadIdx.x;
  if (j >= 2 * BATCH) return;
  int k = kings[(j >> 1) * 2 + (j & 1)];
  int pos = ((int*)(ws + OFF_OFFS))[k] + atomicAdd((int*)(ws + OFF_CUR) + k, 1);
  ((unsigned*)(ws + OFF_JOBS))[pos] = (unsigned)j;
}

__global__ void __launch_bounds__(256) k_feat(unsigned char* __restrict__ ws) {
  const int lane  = threadIdx.x & 31;
  const int wid   = blockIdx.x * (blockDim.x >> 5) + (threadIdx.x >> 5);
  const int tile  = wid >> 4;
  const int ntile = wid & 15;
  const int nTiles = *(const int*)(ws + OFF_NT);
  if (tile >= nTiles) return;
  const int king = ((const int*)(ws + OFF_TK))[tile];
  const int toff = ((const int*)(ws + OFF_TO))[tile];
  const int tcnt = ((const int*)(ws + OFF_TC))[tile];
  const unsigned* jobs = (const unsigned*)(ws + OFF_JOBS);

  const int h16 = lane >> 4;
  const int m   = lane & 15;
  const int nglob = ntile * 16 + m;

  const unsigned myj = jobs[toff + m];
  const unsigned char* arow = (m < tcnt)
      ? (ws + OFF_PP + (size_t)(myj >> 1) * NFEAT)
      : (ws + OFF_ZROW);
  const unsigned char* lorow = ws + OFF_WLO + ((size_t)(king * 256 + nglob)) * NFEAT;
  const unsigned char* hirow = ws + OFF_WHI + ((size_t)(king * 256 + nglob)) * NFEAT;

  v8i clo = {}; v8i chi = {};
  for (int kk = 0; kk < 10; ++kk) {
    const int K0 = kk * 64;
    v2i a0 = *(const v2i*)(arow + K0 +      h16 * 16);
    v2i a1 = *(const v2i*)(arow + K0 + 8  + h16 * 16);
    v2i a2 = *(const v2i*)(arow + K0 + 32 + h16 * 16);
    v2i a3 = *(const v2i*)(arow + K0 + 40 + h16 * 16);
    v8i a = {a0[0], a0[1], a1[0], a1[1], a2[0], a2[1], a3[0], a3[1]};
    v4i l0 = *(const v4i*)(lorow + K0 +      h16 * 16);
    v4i l1 = *(const v4i*)(lorow + K0 + 32 + h16 * 16);
    v4i h0 = *(const v4i*)(hirow + K0 +      h16 * 16);
    v4i h1 = *(const v4i*)(hirow + K0 + 32 + h16 * 16);
    v8i blo = {l0[0], l0[1], l0[2], l0[3], l1[0], l1[1], l1[2], l1[3]};
    v8i bhi = {h0[0], h0[1], h0[2], h0[3], h1[0], h1[1], h1[2], h1[3]};
    if (kk < 9) {
      __builtin_prefetch(lorow + K0 + 64, 0, 3);
      __builtin_prefetch(hirow + K0 + 64, 0, 3);
    }
    clo = __builtin_amdgcn_wmma_i32_16x16x64_iu8(false, a, false, blo, clo, false, false);
    chi = __builtin_amdgcn_wmma_i32_16x16x64_iu8(false, a, true,  bhi, chi, false, false);
  }

  int* res  = (int*)(ws + OFF_RES);
  int* dump = (int*)(ws + OFF_DUMP);
#pragma unroll
  for (int v = 0; v < 8; ++v) {
    int mrow = v + 8 * h16;
    unsigned j = jobs[toff + mrow];
    int b = (int)(j >> 1), h = (int)(j & 1);
    int* dst = (mrow < tcnt) ? (res + h * BATCH * NDIM + b * NDIM + nglob)
                             : (dump + v * 32 + lane);
    *dst = (chi[v] << 8) + clo[v];
  }
}

__global__ void k_act(const int* __restrict__ W, const int* __restrict__ kings,
                      const int* __restrict__ bias, unsigned char* __restrict__ ws) {
  int idx = blockIdx.x * blockDim.x + threadIdx.x;
  int b = idx >> 8, d = idx & 255;
  const int* res = (const int*)(ws + OFF_RES);
  int k0 = kings[b * 2], k1 = kings[b * 2 + 1];
  int acc = res[b * NDIM + d] + res[BATCH * NDIM + b * NDIM + d]
          + W[k0 * KTAB + 640 * NDIM + d] + W[k1 * KTAB + 640 * NDIM + d] + bias[d];
  short s16 = (short)acc;
  int x = s16; x = x < 0 ? 0 : (x > 127 ? 127 : x);
  unsigned char* X = ws + OFF_X;
  X[b * 512 + d] = (unsigned char)x;
  X[b * 512 + 256 + d] = (unsigned char)x;
}

__global__ void __launch_bounds__(128) k_fc1(const int* __restrict__ b1,
                                             unsigned char* __restrict__ ws) {
  int lane = threadIdx.x & 31;
  int wid = blockIdx.x * (blockDim.x >> 5) + (threadIdx.x >> 5);
  int mtile = wid >> 1, ntile = wid & 1;
  int h16 = lane >> 4, m = lane & 15;
  const unsigned char* xrow = ws + OFF_X + (size_t)(mtile * 16 + m) * 512;
  const unsigned char* brow = ws + OFF_W1P + (size_t)(ntile * 16 + m) * 512;
  int nglob = ntile * 16 + m;
  v8i c = {};
  for (int kk = 0; kk < 8; ++kk) {
    int K0 = kk * 64;
    v2i a0 = *(const v2i*)(xrow + K0 +      h16 * 16);
    v2i a1 = *(const v2i*)(xrow + K0 + 8  + h16 * 16);
    v2i a2 = *(const v2i*)(xrow + K0 + 32 + h16 * 16);
    v2i a3 = *(const v2i*)(xrow + K0 + 40 + h16 * 16);
    v8i a = {a0[0], a0[1], a1[0], a1[1], a2[0], a2[1], a3[0], a3[1]};
    v4i b0 = *(const v4i*)(brow + K0 +      h16 * 16);
    v4i b1v = *(const v4i*)(brow + K0 + 32 + h16 * 16);
    v8i bf = {b0[0], b0[1], b0[2], b0[3], b1v[0], b1v[1], b1v[2], b1v[3]};
    c = __builtin_amdgcn_wmma_i32_16x16x64_iu8(false, a, true, bf, c, false, false);
  }
  unsigned char* X1 = ws + OFF_X1;
  int bn = b1[nglob];
#pragma unroll
  for (int v = 0; v < 8; ++v) {
    int mrow = mtile * 16 + v + 8 * h16;
    int y = (c[v] + bn) >> 6;
    y = y < 0 ? 0 : (y > 127 ? 127 : y);
    X1[mrow * 32 + nglob] = (unsigned char)y;
  }
}

__global__ void __launch_bounds__(128) k_fc2(const int* __restrict__ b2,
                                             unsigned char* __restrict__ ws) {
  int lane = threadIdx.x & 31;
  int wid = blockIdx.x * (blockDim.x >> 5) + (threadIdx.x >> 5);
  int mtile = wid >> 1, ntile = wid & 1;
  int h16 = lane >> 4, m = lane & 15;
  const unsigned char* xrow = ws + OFF_X1 + (size_t)(mtile * 16 + m) * 32;
  const unsigned char* brow = ws + OFF_W2P + (size_t)(ntile * 16 + m) * 64;
  const char* owp = (const char*)(ws + OFF_OWP);
  int nglob = ntile * 16 + m;
  v2i a0 = *(const v2i*)(xrow +      h16 * 16);
  v2i a1 = *(const v2i*)(xrow + 8  + h16 * 16);
  v8i a = {a0[0], a0[1], a1[0], a1[1], 0, 0, 0, 0};
  v4i b0 = *(const v4i*)(brow +      h16 * 16);
  v4i b1v = *(const v4i*)(brow + 32 + h16 * 16);
  v8i bf = {b0[0], b0[1], b0[2], b0[3], b1v[0], b1v[1], b1v[2], b1v[3]};
  v8i c = {};
  c = __builtin_amdgcn_wmma_i32_16x16x64_iu8(false, a, true, bf, c, false, false);
  int bn = b2[nglob];
  int ow = (int)owp[nglob];
  int part = 0;
#pragma unroll
  for (int v = 0; v < 8; ++v) {
    int y = (c[v] + bn) >> 6;
    y = y < 0 ? 0 : (y > 127 ? 127 : y);
    part += y * ow;
  }
  atomicAdd((int*)(ws + OFF_PSUM), part);
}

__global__ void k_final(const int* __restrict__ outb, unsigned char* __restrict__ ws,
                        float* __restrict__ out) {
  if (threadIdx.x != 0 || blockIdx.x != 0) return;
  int s = *(const int*)(ws + OFF_PSUM) + outb[0];
  out[0] = (float)(s >> 4);
}

extern "C" void kernel_launch(void* const* d_in, const int* in_sizes, int n_in,
                              void* d_out, int out_size, void* d_ws, size_t ws_size,
                              hipStream_t stream) {
  const int* pp    = (const int*)d_in[0];
  const int* kings = (const int*)d_in[1];
  const int* W     = (const int*)d_in[2];
  const int* bias  = (const int*)d_in[3];
  const int* w1    = (const int*)d_in[4];
  const int* b1    = (const int*)d_in[5];
  const int* w2    = (const int*)d_in[6];
  const int* b2    = (const int*)d_in[7];
  const int* ow    = (const int*)d_in[8];
  const int* outb  = (const int*)d_in[9];
  unsigned char* ws = (unsigned char*)d_ws;
  float* out = (float*)d_out;

  k_prep<<<2560, 256, 0, stream>>>(pp, w1, w2, ow, ws);
  k_wsplit<<<(64 * 256 * 640) / 256, 256, 0, stream>>>(W, ws);
  k_count<<<8, 256, 0, stream>>>(kings, ws);
  k_scan<<<1, 32, 0, stream>>>(ws);
  k_scatter<<<8, 256, 0, stream>>>(kings, ws);
  k_feat<<<384, 256, 0, stream>>>(ws);
  k_act<<<(BATCH * NDIM) / 256, 256, 0, stream>>>(W, kings, bias, ws);
  k_fc1<<<32, 128, 0, stream>>>(b1, ws);
  k_fc2<<<32, 128, 0, stream>>>(b2, ws);
  k_final<<<1, 32, 0, stream>>>(outb, ws, out);
}
